// EchoPIPs_33895881900437
// MI455X (gfx1250) — matrix operand order repaired and checked by reading, not compiled, not run on hardware
//
#include <hip/hip_runtime.h>
#include <hip/hip_bf16.h>

typedef __attribute__((ext_vector_type(16))) _Float16 v16h;
typedef __attribute__((ext_vector_type(8)))  _Float16 h8_t;
typedef __attribute__((ext_vector_type(8)))  float    v8f;

#define S_    16
#define NPTS  64
#define HIMG  200
#define WIMG  240
#define LAT_  64
#define HID_  128
#define RAD_  3
#define LVL_  4
#define FFD_  3000
#define ITERS_ 4
#define FCW_  608   // fcorr_raw padded width (588 -> 608, multiple of 32)

__device__ __forceinline__ int imin(int a, int b) { return a < b ? a : b; }
__device__ __forceinline__ int imax(int a, int b) { return a > b ? a : b; }

// ---------------------------------------------------------------------------
__global__ void k_normalize(const float* __restrict__ in, float* __restrict__ out, int n) {
    int i = blockIdx.x * blockDim.x + threadIdx.x;
    if (i < n) out[i] = 2.0f * (in[i] * (1.0f / 255.0f)) - 1.0f;
}

__global__ void k_zero_h(_Float16* __restrict__ p, int n) {
    int i = blockIdx.x * blockDim.x + threadIdx.x;
    if (i < n) p[i] = (_Float16)0.0f;
}

// ---------------------------------------------------------------------------
// frame_flow: [S,200,240] (frame0 = last frame, else temporal diff) resized to
// [S,50,60], align_corners=False, fused.
__global__ void k_frame_flow(const float* __restrict__ rgbs_n, float* __restrict__ ff) {
    int i = blockIdx.x * blockDim.x + threadIdx.x;
    if (i >= S_ * 50 * 60) return;
    int f = i / 3000; int r = i % 3000; int oy = r / 60, ox = r % 60;
    float sy = fminf(fmaxf((oy + 0.5f) * (HIMG / 50.0f) - 0.5f, 0.0f), HIMG - 1.0f);
    float sx = fminf(fmaxf((ox + 0.5f) * (WIMG / 60.0f) - 0.5f, 0.0f), WIMG - 1.0f);
    int y0 = (int)floorf(sy); int y1 = imin(y0 + 1, HIMG - 1); float wy = sy - (float)y0;
    int x0 = (int)floorf(sx); int x1 = imin(x0 + 1, WIMG - 1); float wx = sx - (float)x0;
    auto src = [&](int y, int x) -> float {
        int idx = y * WIMG + x;
        if (f == 0) return rgbs_n[(S_ - 1) * HIMG * WIMG + idx];
        return rgbs_n[f * HIMG * WIMG + idx] - rgbs_n[(f - 1) * HIMG * WIMG + idx];
    };
    ff[i] = src(y0, x0) * (1.0f - wy) * (1.0f - wx) + src(y0, x1) * (1.0f - wy) * wx
          + src(y1, x0) * wy * (1.0f - wx) + src(y1, x1) * wy * wx;
}

// ---------------------------------------------------------------------------
// patchify conv: in [S,1,200,240], w [64,1,k,k], stride==k -> [S,64,OH,OW]
__global__ void k_conv(const float* __restrict__ in, const float* __restrict__ w,
                       float* __restrict__ out, int ksz, int stride, int OH, int OW) {
    int i = blockIdx.x * blockDim.x + threadIdx.x;
    int tot = S_ * LAT_ * OH * OW;
    if (i >= tot) return;
    int ox = i % OW; int t = i / OW; int oy = t % OH; t /= OH; int oc = t % LAT_; int s = t / LAT_;
    const float* ip = in + (size_t)s * HIMG * WIMG + (size_t)oy * stride * WIMG + (size_t)ox * stride;
    const float* wp = w + (size_t)oc * ksz * ksz;
    float acc = 0.0f;
    for (int ky = 0; ky < ksz; ++ky)
        for (int kx = 0; kx < ksz; ++kx)
            acc += ip[ky * WIMG + kx] * wp[ky * ksz + kx];
    out[i] = acc;
}

// ---------------------------------------------------------------------------
// instance norm over HW per (q,c): one block per (q,c), LDS tree reduction
__global__ void k_instnorm(float* __restrict__ x, int HW) {
    __shared__ float ssum[256];
    __shared__ float ssq[256];
    int qc = blockIdx.x;
    float* p = x + (size_t)qc * HW;
    float s = 0.0f, sq = 0.0f;
    for (int i = threadIdx.x; i < HW; i += blockDim.x) { float v = p[i]; s += v; sq += v * v; }
    ssum[threadIdx.x] = s; ssq[threadIdx.x] = sq;
    __syncthreads();
    for (int st = blockDim.x / 2; st > 0; st >>= 1) {
        if ((int)threadIdx.x < st) { ssum[threadIdx.x] += ssum[threadIdx.x + st]; ssq[threadIdx.x] += ssq[threadIdx.x + st]; }
        __syncthreads();
    }
    float m = ssum[0] / (float)HW;
    float var = ssq[0] / (float)HW - m * m;
    float inv = rsqrtf(var + 1e-5f);
    for (int i = threadIdx.x; i < HW; i += blockDim.x) p[i] = (p[i] - m) * inv;
}

// ---------------------------------------------------------------------------
__global__ void k_avgpool(const float* __restrict__ in, float* __restrict__ out,
                          int QC, int H, int W) {
    int OH = H / 2, OW = W / 2;
    int i = blockIdx.x * blockDim.x + threadIdx.x;
    if (i >= QC * OH * OW) return;
    int ox = i % OW; int t = i / OW; int oy = t % OH; int qc = t / OH;
    const float* p = in + ((size_t)qc * H + (size_t)oy * 2) * W + (size_t)ox * 2;
    out[i] = 0.25f * (p[0] + p[1] + p[W] + p[W + 1]);
}

// ---------------------------------------------------------------------------
// bilinear sample with zero-outside border (matches reference bilinear_sample)
__device__ __forceinline__ float bsample(const float* __restrict__ img, int H, int W,
                                         float x, float y) {
    float fx = floorf(x), fy = floorf(y);
    int x0 = (int)fx, y0 = (int)fy;
    float wx = x - fx, wy = y - fy;
    float v00 = (x0 >= 0 && x0 < W && y0 >= 0 && y0 < H) ? img[y0 * W + x0] : 0.0f;
    float v01 = (x0 + 1 >= 0 && x0 + 1 < W && y0 >= 0 && y0 < H) ? img[y0 * W + x0 + 1] : 0.0f;
    float v10 = (x0 >= 0 && x0 < W && y0 + 1 >= 0 && y0 + 1 < H) ? img[(y0 + 1) * W + x0] : 0.0f;
    float v11 = (x0 + 1 >= 0 && x0 + 1 < W && y0 + 1 >= 0 && y0 + 1 < H) ? img[(y0 + 1) * W + x0 + 1] : 0.0f;
    return v00 * (1.0f - wy) * (1.0f - wx) + v01 * (1.0f - wy) * wx
         + v10 * wy * (1.0f - wx) + v11 * wy * wx;
}

// f0[n][c] = bilinear(fmap frame0 channel c, coords[s=0,n])  (f16: GEMM A operand)
__global__ void k_sample_f0(const float* __restrict__ fmap0, const float* __restrict__ coords,
                            _Float16* __restrict__ f0, int H, int W) {
    int i = blockIdx.x * blockDim.x + threadIdx.x;
    if (i >= NPTS * LAT_) return;
    int c = i % LAT_, n = i / LAT_;
    float x = coords[n * 2 + 0], y = coords[n * 2 + 1];
    f0[n * LAT_ + c] = (_Float16)bsample(fmap0 + (size_t)c * H * W, H, W, x, y);
}

// feats[s][n][c] from frame max(s-shift,0) at that frame's coords (f16)
__global__ void k_sample_shift(const float* __restrict__ fmaps, const float* __restrict__ coords,
                               _Float16* __restrict__ feats, int H, int W, int shift) {
    int i = blockIdx.x * blockDim.x + threadIdx.x;
    if (i >= S_ * NPTS * LAT_) return;
    int c = i % LAT_; int t = i / LAT_; int n = t % NPTS; int s = t / NPTS;
    int qs = imax(s - shift, 0);
    float x = coords[(qs * NPTS + n) * 2 + 0], y = coords[(qs * NPTS + n) * 2 + 1];
    feats[((size_t)s * NPTS + n) * LAT_ + c] =
        (_Float16)bsample(fmaps + ((size_t)qs * LAT_ + c) * H * W, H, W, x, y);
}

// ---------------------------------------------------------------------------
// Pack B [K x Nd] (f32 row-major, ldb) into WMMA lane layout, f16, zero-padded:
// Bp[((kt*ntiles + nt)*32 + lane)*16 + e] = B[(kt*32 + (e>>3)*16 + (lane>>4)*8 + (e&7))*ldb + nt*16 + (lane&15)]
// Each GEMM wave then loads its 16 B-halves as one contiguous 32B v16h.
__global__ void k_pack_b(const float* __restrict__ B, long long sBq,
                         _Float16* __restrict__ Bp, long long sBpq,
                         int K, int Nd, int ldb, int ntiles, int tot) {
    int i = blockIdx.x * blockDim.x + threadIdx.x;
    if (i >= tot) return;
    int q = blockIdx.z;
    B += (size_t)q * sBq; Bp += (size_t)q * sBpq;
    int e = i & 15; int lane = (i >> 4) & 31; int tile = i >> 9;
    int nt = tile % ntiles; int kt = tile / ntiles;
    int k = kt * 32 + (e >> 3) * 16 + (lane >> 4) * 8 + (e & 7);
    int col = nt * 16 + (lane & 15);
    Bp[i] = (k < K && col < Nd) ? (_Float16)B[(size_t)k * ldb + col] : (_Float16)0.0f;
}

// ---------------------------------------------------------------------------
// Batched WMMA GEMM with pre-packed B. A f16 row-major (K padded to 32*Kt,
// zero-filled), Bp in WMMA lane layout. 4 waves/block, one 16x16 tile/wave.
// Inner loop: 2x b128 A loads + 2x b128 B loads + v_wmma_f32_16x16x32_f16.
__global__ void k_wmma_gemm(const _Float16* __restrict__ A, long long sAq,
                            const _Float16* __restrict__ Bp, long long sBpq,
                            const float* __restrict__ bias,
                            float* __restrict__ C, long long sCq,
                            int M, int Kt, int Nd, int lda, int ldc, int ntiles,
                            float scale, int relu) {
    int q = blockIdx.z;
    A += (size_t)q * sAq; Bp += (size_t)q * sBpq; C += (size_t)q * sCq;
    int wave = threadIdx.x >> 5;
    int nt = blockIdx.x * (blockDim.x >> 5) + wave;
    if (nt >= ntiles) return;
    int n0 = nt * 16;
    int m0 = blockIdx.y * 16;
    int lane = threadIdx.x & 31;
    int half = lane >> 4;
    int r    = lane & 15;
    const _Float16* Arow = A + (size_t)(m0 + r) * lda + half * 8;
    const _Float16* bptr = Bp + ((size_t)nt * 32 + lane) * 16;
    size_t bstep = (size_t)ntiles * 512;

    v8f acc = {0.f, 0.f, 0.f, 0.f, 0.f, 0.f, 0.f, 0.f};
    #pragma unroll 2
    for (int kt = 0; kt < Kt; ++kt) {
        h8_t a0 = *(const h8_t*)(Arow);        // K: half*8 + 0..7
        h8_t a1 = *(const h8_t*)(Arow + 16);   // K: 16 + half*8 + 0..7
        v16h b  = *(const v16h*)(bptr);        // packed: exact lane layout
        v16h a;
        #pragma unroll
        for (int e = 0; e < 8; ++e) { a[e] = a0[e]; a[8 + e] = a1[e]; }
        acc = __builtin_amdgcn_wmma_f32_16x16x32_f16(false, a, false, b, (short)0, acc,
                                                     false, false);
        Arow += 32; bptr += bstep;
    }
    #pragma unroll
    for (int j = 0; j < 8; ++j) {
        int rr = m0 + j + 8 * half;   // C/D layout: vgpr j -> M=j (lanes 0-15) / M=j+8
        int cc = n0 + r;
        if (cc < Nd) {
            float v = acc[j] * scale + (bias ? bias[cc] : 0.0f);
            if (relu) v = fmaxf(v, 0.0f);
            C[(size_t)rr * ldc + cc] = v;
        }
    }
}

// ---------------------------------------------------------------------------
// resize corr level [S,N,H,W] -> 14x14 (align_corners=True); mean across levels
__global__ void k_resize14_acc(const float* __restrict__ corr, float* __restrict__ fc,
                               int H, int W, float wsum, int zero_first) {
    int i = blockIdx.x * blockDim.x + threadIdx.x;
    if (i >= S_ * NPTS * 196) return;
    int p = i % 196; int t = i / 196; int n = t % NPTS; int s = t / NPTS;
    int py = p / 14, px = p % 14;
    float sy = (float)py * (float)(H - 1) / 13.0f;
    float sx = (float)px * (float)(W - 1) / 13.0f;
    int y0 = (int)floorf(sy); int y1 = imin(y0 + 1, H - 1); float wy = sy - (float)y0;
    int x0 = (int)floorf(sx); int x1 = imin(x0 + 1, W - 1); float wx = sx - (float)x0;
    const float* pl = corr + ((size_t)s * NPTS + n) * H * W;
    float v = pl[y0 * W + x0] * (1.0f - wy) * (1.0f - wx) + pl[y0 * W + x1] * (1.0f - wy) * wx
            + pl[y1 * W + x0] * wy * (1.0f - wx) + pl[y1 * W + x1] * wy * wx;
    float* dst = &fc[((size_t)n * S_ + s) * 196 + p];
    if (zero_first) *dst = wsum * v; else *dst += wsum * v;
}

// ---------------------------------------------------------------------------
// corr_sample: 7x7 window around coords/2^lvl -> fcorr f16 [N*S, FCW_]
__global__ void k_corr_sample(const float* __restrict__ corr, const float* __restrict__ coords,
                              _Float16* __restrict__ fcorr, int H, int W, int lvl, int setoff) {
    int i = blockIdx.x * blockDim.x + threadIdx.x;
    if (i >= S_ * NPTS * 49) return;
    int k = i % 49; int t = i / 49; int n = t % NPTS; int s = t / NPTS;
    float inv = 1.0f / (float)(1 << lvl);
    float cx = coords[(s * NPTS + n) * 2 + 0] * inv + (float)(k % 7 - RAD_);
    float cy = coords[(s * NPTS + n) * 2 + 1] * inv + (float)(k / 7 - RAD_);
    float v = bsample(corr + ((size_t)s * NPTS + n) * H * W, H, W, cx, cy);
    fcorr[((size_t)n * S_ + s) * FCW_ + setoff + lvl * 49 + k] = (_Float16)v;
}

// ---------------------------------------------------------------------------
// assemble DeltaBlock input (f16, K padded): [fc(fcw)|flow(2)|frame_flow(3000)|xy0(2)|0..]
__global__ void k_hcat(const float* __restrict__ fc, int fcw,
                       const float* __restrict__ coords, const float* __restrict__ ff,
                       const float* __restrict__ xy0, _Float16* __restrict__ hcat,
                       int D, int Dpad) {
    int i = blockIdx.x * blockDim.x + threadIdx.x;
    int tot = NPTS * S_ * Dpad;
    if (i >= tot) return;
    int col = i % Dpad; int row = i / Dpad; int s = row % S_; int n = row / S_;
    float v;
    if (col < fcw) {
        v = fc[(size_t)row * fcw + col];
    } else if (col < fcw + 2) {
        int d = col - fcw;
        int s2 = imin(s, S_ - 2);  // flows_from duplicates the last diff
        v = coords[((s2 + 1) * NPTS + n) * 2 + d] - coords[(s2 * NPTS + n) * 2 + d];
    } else if (col < fcw + 2 + FFD_) {
        v = ff[s * FFD_ + (col - fcw - 2)];
    } else if (col < D) {
        v = xy0[n * 2 + (col - fcw - 2 - FFD_)];
    } else {
        v = 0.0f;  // K padding
    }
    hcat[i] = (_Float16)v;
}

// ---------------------------------------------------------------------------
// second MLP layer: hidden [rows,128] @ w2 [128,2] + b2 -> delta [rows,2]
__global__ void k_mlp_out(const float* __restrict__ hidden, const float* __restrict__ w2,
                          const float* __restrict__ b2, float* __restrict__ delta) {
    int i = blockIdx.x * blockDim.x + threadIdx.x;
    if (i >= NPTS * S_ * 2) return;
    int d = i % 2; int row = i / 2;
    float acc = b2[d];
    const float* h = hidden + (size_t)row * HID_;
    #pragma unroll 4
    for (int k = 0; k < HID_; ++k) acc += h[k] * w2[k * 2 + d];
    delta[i] = acc;
}

// ---------------------------------------------------------------------------
__global__ void k_init_coords(const float* __restrict__ pts, float* __restrict__ coords) {
    int i = blockIdx.x * blockDim.x + threadIdx.x;
    if (i >= S_ * NPTS * 2) return;
    int d = i % 2; int n = (i / 2) % NPTS;
    coords[i] = pts[n * 2 + d] * 0.125f;
}
__global__ void k_update_coords(float* __restrict__ coords, const float* __restrict__ delta) {
    int i = blockIdx.x * blockDim.x + threadIdx.x;
    if (i >= S_ * NPTS * 2) return;
    int d = i % 2; int t = i / 2; int n = t % NPTS; int s = t / NPTS;
    coords[i] += delta[((size_t)n * S_ + s) * 2 + d];
}
__global__ void k_scale_coords(float* __restrict__ coords, float f) {
    int i = blockIdx.x * blockDim.x + threadIdx.x;
    if (i < S_ * NPTS * 2) coords[i] *= f;
}
__global__ void k_write_preds(const float* __restrict__ coords, float* __restrict__ out, float f) {
    int i = blockIdx.x * blockDim.x + threadIdx.x;
    if (i < S_ * NPTS * 2) out[i] = coords[i] * f;
}
__global__ void k_copy_xy0(const float* __restrict__ coords, float* __restrict__ xy0) {
    int i = blockIdx.x * blockDim.x + threadIdx.x;
    if (i < NPTS * 2) xy0[i] = coords[i];  // s=0 slice is the first N*2 floats
}

// ===========================================================================
extern "C" void kernel_launch(void* const* d_in, const int* in_sizes, int n_in,
                              void* d_out, int out_size, void* d_ws, size_t ws_size,
                              hipStream_t stream) {
    (void)in_sizes; (void)n_in; (void)out_size; (void)ws_size;
    const float* rgbs      = (const float*)d_in[0];
    const float* points0   = (const float*)d_in[1];
    const float* fnet_w    = (const float*)d_in[2];
    const float* fnet_fw   = (const float*)d_in[3];
    const float* linear_w  = (const float*)d_in[4];
    const float* linear_b  = (const float*)d_in[5];
    const float* dbs_w1    = (const float*)d_in[6];
    const float* dbs_b1    = (const float*)d_in[7];
    const float* dbs_w2    = (const float*)d_in[8];
    const float* dbs_b2    = (const float*)d_in[9];
    const float* dbt_w1    = (const float*)d_in[10];
    const float* dbt_b1    = (const float*)d_in[11];
    const float* dbt_w2    = (const float*)d_in[12];
    const float* dbt_b2    = (const float*)d_in[13];
    float* outp            = (float*)d_out;

    // ---- workspace layout (64B aligned allocations) ----
    char* base = (char*)d_ws;
    size_t off = 0;
    auto allocf = [&](size_t n) { float* p = (float*)(base + off); off = (off + n * 4 + 63) & ~(size_t)63; return p; };
    auto alloch = [&](size_t n) { _Float16* p = (_Float16*)(base + off); off = (off + n * 2 + 63) & ~(size_t)63; return p; };

    float* rgbs_n    = allocf((size_t)S_ * HIMG * WIMG);
    float* ffb       = allocf((size_t)S_ * FFD_);
    float* fm_c      = allocf((size_t)S_ * LAT_ * 25 * 30);
    float* pc1       = allocf((size_t)S_ * LAT_ * 12 * 15);
    float* pc2       = allocf((size_t)S_ * LAT_ * 6 * 7);
    float* pc3       = allocf((size_t)S_ * LAT_ * 3 * 3);
    float* cc0       = allocf((size_t)S_ * NPTS * 25 * 30);
    float* cc1       = allocf((size_t)S_ * NPTS * 12 * 15);
    float* cc2       = allocf((size_t)S_ * NPTS * 6 * 7);
    float* cc3       = allocf((size_t)S_ * NPTS * 3 * 3);
    float* fcorrs_c  = allocf((size_t)NPTS * S_ * 196);
    float* coords    = allocf((size_t)S_ * NPTS * 2);
    float* xy0       = allocf((size_t)NPTS * 2);
    float* hidden    = allocf((size_t)NPTS * S_ * HID_);
    float* delta     = allocf((size_t)NPTS * S_ * 2);
    float* fm_f      = allocf((size_t)S_ * LAT_ * 100 * 120);
    float* pf1       = allocf((size_t)S_ * LAT_ * 50 * 60);
    float* pf2       = allocf((size_t)S_ * LAT_ * 25 * 30);
    float* pf3       = allocf((size_t)S_ * LAT_ * 12 * 15);
    float* c1_0      = allocf((size_t)S_ * NPTS * 100 * 120);
    float* c1_1      = allocf((size_t)S_ * NPTS * 50 * 60);
    float* c1_2      = allocf((size_t)S_ * NPTS * 25 * 30);
    float* c1_3      = allocf((size_t)S_ * NPTS * 12 * 15);
    float* cX0       = allocf((size_t)S_ * NPTS * 100 * 120);
    float* cX1       = allocf((size_t)S_ * NPTS * 50 * 60);
    float* cX2       = allocf((size_t)S_ * NPTS * 25 * 30);
    float* cX3       = allocf((size_t)S_ * NPTS * 12 * 15);
    float* fcorr_lin = allocf((size_t)NPTS * S_ * 392);
    float* xy0f      = allocf((size_t)NPTS * 2);

    // ntile counts per level (N = Hl*Wl of each corr GEMM)
    const int Hc[4] = {25, 12, 6, 3},    Wc[4] = {30, 15, 7, 3};
    const int Hf[4] = {100, 50, 25, 12}, Wf[4] = {120, 60, 30, 15};
    int ntc[4], ntf[4];
    for (int l = 0; l < 4; ++l) { ntc[l] = (Hc[l] * Wc[l] + 15) / 16; ntf[l] = (Hf[l] * Wf[l] + 15) / 16; }

    // f16 A-operand buffers
    _Float16* hcat_h   = alloch((size_t)NPTS * S_ * 3424);
    _Float16* fcraw_h  = alloch((size_t)NPTS * S_ * FCW_);
    _Float16* f0c_h    = alloch((size_t)NPTS * LAT_);
    _Float16* f0f_h    = alloch((size_t)NPTS * LAT_);
    _Float16* featsX_h = alloch((size_t)S_ * NPTS * LAT_);
    // packed B buffers (per-q size = Kt*ntiles*512 halves; corr: Kt=2)
    _Float16* pcpk[4]; _Float16* pfpk[4];
    for (int l = 0; l < 4; ++l) pcpk[l] = alloch((size_t)S_ * 2 * ntc[l] * 512);
    for (int l = 0; l < 4; ++l) pfpk[l] = alloch((size_t)S_ * 2 * ntf[l] * 512);
    _Float16* dbs_w1_pk = alloch((size_t)100 * 8 * 512);   // K=3200 -> Kt=100, N=128 -> 8 nt
    _Float16* dbt_w1_pk = alloch((size_t)107 * 8 * 512);   // K=3396 -> Kt=107 (zero-padded)
    _Float16* lin_w_pk  = alloch((size_t)19 * 25 * 512);   // K=588 -> Kt=19, N=392 -> 25 nt

    auto blocks = [](long long n, int bs) { return dim3((unsigned)((n + bs - 1) / bs)); };
    const int BS = 256;

    float*    pcs[4] = {fm_c, pc1, pc2, pc3};
    float*    ccs[4] = {cc0, cc1, cc2, cc3};
    float*    pfs[4] = {fm_f, pf1, pf2, pf3};
    float*    c1s[4] = {c1_0, c1_1, c1_2, c1_3};
    float*    cXs[4] = {cX0, cX1, cX2, cX3};
    const float cscale = 0.125f;  // 1/sqrt(64)

    auto pack_b = [&](const float* B, long long sBq, _Float16* Bp, long long sBpq,
                      int K, int Nd, int ldb, int ntiles, int Kt, int batch) {
        int tot = Kt * ntiles * 512;
        dim3 g = blocks(tot, BS); g.z = batch;
        k_pack_b<<<g, BS, 0, stream>>>(B, sBq, Bp, sBpq, K, Nd, ldb, ntiles, tot);
    };
    auto gemm = [&](const _Float16* A, long long sAq, const _Float16* Bp, long long sBpq,
                    const float* bias, float* Cm, long long sCq,
                    int M, int Kt, int Nd, int lda, int ldc, int ntiles,
                    float scale, int relu, int batch) {
        dim3 g((unsigned)((ntiles + 3) / 4), (unsigned)(M / 16), (unsigned)batch);
        k_wmma_gemm<<<g, dim3(128), 0, stream>>>(A, sAq, Bp, sBpq, bias, Cm, sCq,
                                                 M, Kt, Nd, lda, ldc, ntiles, scale, relu);
    };

    // ---------- preprocessing ----------
    long long nrg = (long long)S_ * HIMG * WIMG;
    k_normalize<<<blocks(nrg, BS), BS, 0, stream>>>(rgbs, rgbs_n, (int)nrg);
    k_frame_flow<<<blocks((long long)S_ * 50 * 60, BS), BS, 0, stream>>>(rgbs_n, ffb);
    k_zero_h<<<blocks((long long)NPTS * S_ * FCW_, BS), BS, 0, stream>>>(fcraw_h, NPTS * S_ * FCW_);

    // weights -> packed f16 (K zero-padded inside pack)
    pack_b(dbs_w1, 0, dbs_w1_pk, 0, 3200, HID_, HID_, 8, 100, 1);
    pack_b(dbt_w1, 0, dbt_w1_pk, 0, 3396, HID_, HID_, 8, 107, 1);
    pack_b(linear_w, 0, lin_w_pk, 0, 588, 392, 392, 25, 19, 1);

    // ---------- coarse stage (stride 8) ----------
    k_conv<<<blocks((long long)S_ * LAT_ * 25 * 30, BS), BS, 0, stream>>>(rgbs_n, fnet_w, fm_c, 8, 8, 25, 30);
    k_instnorm<<<dim3(S_ * LAT_), dim3(256), 0, stream>>>(fm_c, 25 * 30);
    k_avgpool<<<blocks((long long)S_ * LAT_ * 12 * 15, BS), BS, 0, stream>>>(fm_c, pc1, S_ * LAT_, 25, 30);
    k_avgpool<<<blocks((long long)S_ * LAT_ * 6 * 7, BS), BS, 0, stream>>>(pc1, pc2, S_ * LAT_, 12, 15);
    k_avgpool<<<blocks((long long)S_ * LAT_ * 3 * 3, BS), BS, 0, stream>>>(pc2, pc3, S_ * LAT_, 6, 7);
    for (int l = 0; l < 4; ++l) {
        int HW = Hc[l] * Wc[l];
        pack_b(pcs[l], (long long)LAT_ * HW, pcpk[l], (long long)2 * ntc[l] * 512,
               LAT_, HW, HW, ntc[l], 2, S_);
    }

    k_init_coords<<<blocks(S_ * NPTS * 2, BS), BS, 0, stream>>>(points0, coords);
    k_copy_xy0<<<blocks(NPTS * 2, BS), BS, 0, stream>>>(coords, xy0);
    k_sample_f0<<<blocks(NPTS * LAT_, BS), BS, 0, stream>>>(fm_c, coords, f0c_h, 25, 30);

    for (int l = 0; l < 4; ++l) {
        int HW = Hc[l] * Wc[l];
        gemm(f0c_h, 0, pcpk[l], (long long)2 * ntc[l] * 512, nullptr, ccs[l], (long long)NPTS * HW,
             NPTS, 2, HW, LAT_, HW, ntc[l], cscale, 0, S_);
    }
    for (int l = 0; l < 4; ++l)
        k_resize14_acc<<<blocks((long long)S_ * NPTS * 196, BS), BS, 0, stream>>>(
            ccs[l], fcorrs_c, Hc[l], Wc[l], 0.25f, l == 0 ? 1 : 0);

    // coarse DeltaBlock (D = 196+2+3000+2 = 3200, already multiple of 32)
    k_hcat<<<blocks((long long)NPTS * S_ * 3200, BS), BS, 0, stream>>>(fcorrs_c, 196, coords, ffb, xy0, hcat_h, 3200, 3200);
    gemm(hcat_h, 0, dbs_w1_pk, 0, dbs_b1, hidden, 0, NPTS * S_, 100, HID_, 3200, HID_, 8, 1.0f, 1, 1);
    k_mlp_out<<<blocks(NPTS * S_ * 2, BS), BS, 0, stream>>>(hidden, dbs_w2, dbs_b2, delta);
    k_update_coords<<<blocks(S_ * NPTS * 2, BS), BS, 0, stream>>>(coords, delta);
    k_write_preds<<<blocks(S_ * NPTS * 2, BS), BS, 0, stream>>>(coords, outp, 8.0f);  // preds[0]

    // ---------- fine stage (stride 2) ----------
    k_scale_coords<<<blocks(S_ * NPTS * 2, BS), BS, 0, stream>>>(coords, 4.0f);  // coords_init/2
    k_conv<<<blocks((long long)S_ * LAT_ * 100 * 120, BS), BS, 0, stream>>>(rgbs_n, fnet_fw, fm_f, 2, 2, 100, 120);
    k_instnorm<<<dim3(S_ * LAT_), dim3(256), 0, stream>>>(fm_f, 100 * 120);
    k_avgpool<<<blocks((long long)S_ * LAT_ * 50 * 60, BS), BS, 0, stream>>>(fm_f, pf1, S_ * LAT_, 100, 120);
    k_avgpool<<<blocks((long long)S_ * LAT_ * 25 * 30, BS), BS, 0, stream>>>(pf1, pf2, S_ * LAT_, 50, 60);
    k_avgpool<<<blocks((long long)S_ * LAT_ * 12 * 15, BS), BS, 0, stream>>>(pf2, pf3, S_ * LAT_, 25, 30);
    for (int l = 0; l < 4; ++l) {
        int HW = Hf[l] * Wf[l];
        pack_b(pfs[l], (long long)LAT_ * HW, pfpk[l], (long long)2 * ntf[l] * 512,
               LAT_, HW, HW, ntf[l], 2, S_);
    }

    k_copy_xy0<<<blocks(NPTS * 2, BS), BS, 0, stream>>>(coords, xy0f);
    k_sample_f0<<<blocks(NPTS * LAT_, BS), BS, 0, stream>>>(fm_f, coords, f0f_h, 100, 120);
    for (int l = 0; l < 4; ++l) {
        int HW = Hf[l] * Wf[l];
        gemm(f0f_h, 0, pfpk[l], (long long)2 * ntf[l] * 512, nullptr, c1s[l], (long long)NPTS * HW,
             NPTS, 2, HW, LAT_, HW, ntf[l], cscale, 0, S_);
    }

    for (int it = 0; it < ITERS_; ++it) {
        // set 0: sample corrs1
        for (int l = 0; l < 4; ++l)
            k_corr_sample<<<blocks((long long)S_ * NPTS * 49, BS), BS, 0, stream>>>(
                c1s[l], coords, fcraw_h, Hf[l], Wf[l], l, 0);
        // set 1: corrs2 (shift 2)
        k_sample_shift<<<blocks((long long)S_ * NPTS * LAT_, BS), BS, 0, stream>>>(fm_f, coords, featsX_h, 100, 120, 2);
        for (int l = 0; l < 4; ++l) {
            int HW = Hf[l] * Wf[l];
            gemm(featsX_h, (long long)NPTS * LAT_, pfpk[l], (long long)2 * ntf[l] * 512, nullptr,
                 cXs[l], (long long)NPTS * HW, NPTS, 2, HW, LAT_, HW, ntf[l], cscale, 0, S_);
        }
        for (int l = 0; l < 4; ++l)
            k_corr_sample<<<blocks((long long)S_ * NPTS * 49, BS), BS, 0, stream>>>(
                cXs[l], coords, fcraw_h, Hf[l], Wf[l], l, 196);
        // set 2: corrs4 (shift 4), reuse cX buffers
        k_sample_shift<<<blocks((long long)S_ * NPTS * LAT_, BS), BS, 0, stream>>>(fm_f, coords, featsX_h, 100, 120, 4);
        for (int l = 0; l < 4; ++l) {
            int HW = Hf[l] * Wf[l];
            gemm(featsX_h, (long long)NPTS * LAT_, pfpk[l], (long long)2 * ntf[l] * 512, nullptr,
                 cXs[l], (long long)NPTS * HW, NPTS, 2, HW, LAT_, HW, ntf[l], cscale, 0, S_);
        }
        for (int l = 0; l < 4; ++l)
            k_corr_sample<<<blocks((long long)S_ * NPTS * 49, BS), BS, 0, stream>>>(
                cXs[l], coords, fcraw_h, Hf[l], Wf[l], l, 392);

        // linear 588 -> 392 (WMMA; A width FCW_=608, zero-padded)
        gemm(fcraw_h, 0, lin_w_pk, 0, linear_b, fcorr_lin, 0, NPTS * S_, 19, 392, FCW_, 392, 25, 1.0f, 0, 1);
        // fine DeltaBlock (D = 392+2+3000+2 = 3396, padded to 3424)
        k_hcat<<<blocks((long long)NPTS * S_ * 3424, BS), BS, 0, stream>>>(fcorr_lin, 392, coords, ffb, xy0f, hcat_h, 3396, 3424);
        gemm(hcat_h, 0, dbt_w1_pk, 0, dbt_b1, hidden, 0, NPTS * S_, 107, HID_, 3424, HID_, 8, 1.0f, 1, 1);
        k_mlp_out<<<blocks(NPTS * S_ * 2, BS), BS, 0, stream>>>(hidden, dbt_w2, dbt_b2, delta);
        k_update_coords<<<blocks(S_ * NPTS * 2, BS), BS, 0, stream>>>(coords, delta);
        k_write_preds<<<blocks(S_ * NPTS * 2, BS), BS, 0, stream>>>(
            coords, outp + (size_t)(it + 1) * S_ * NPTS * 2, 2.0f);  // preds[it+1]
    }
}
